// TransformerLayer_65481071394964
// MI455X (gfx1250) — matrix operand order repaired and checked by reading, not compiled, not run on hardware
//
#include <hip/hip_runtime.h>
#include <cstddef>
#include <cstdint>

// ---------------------------------------------------------------------------
// CDNA5 (gfx1250) wave32 WMMA types
// ---------------------------------------------------------------------------
typedef __attribute__((ext_vector_type(16))) __bf16 v16bf;
typedef __attribute__((ext_vector_type(8)))  float  v8f;
typedef __attribute__((ext_vector_type(4)))  unsigned int v4u;
typedef __attribute__((ext_vector_type(8)))  int v8i;
typedef __attribute__((ext_vector_type(4)))  int v4i;

#define WMMA_BF16(a, b, c) \
  __builtin_amdgcn_wmma_f32_16x16x32_bf16(false, (a), false, (b), (short)0, (c), false, false)

#if __has_builtin(__builtin_amdgcn_tensor_load_to_lds) && \
    __has_builtin(__builtin_amdgcn_s_wait_tensorcnt)
#define HAVE_TDM 1
#endif

static __device__ inline v8f vzero8() {
  v8f z;
#pragma unroll
  for (int i = 0; i < 8; ++i) z[i] = 0.0f;
  return z;
}

static __device__ inline __bf16 f2bf(float x) { return (__bf16)x; }

static __device__ inline float sigmoidf_(float x) { return 1.0f / (1.0f + __expf(-x)); }

// ---------------------------------------------------------------------------
// Fragment loaders (wave32, 16x16x32 bf16 WMMA; A and B share a contiguous k order, C/D per ISA 7.12.2)
// A (16x32, MxK): lane L: row r=L&15, half=L>>4; elem i<8 -> k=16*half+i,
//                 elem 8+i -> k=16*half+8+i
// B (32x16, KxN) stored as W row-major [n][k]: lane L: col n=L&15, half=L>>4;
//                 elem i -> k=16*half+i
// C/D (16x16):   vgpr a, lane L: row m=a+8*(L>>4), col n=L&15
// ---------------------------------------------------------------------------
static __device__ inline v16bf frag_A_bf16(const __bf16* t, int ld) {
  int lane = threadIdx.x & 31, r = lane & 15, half = lane >> 4;
  const __bf16* row = t + (size_t)r * ld;
  v16bf a;
#pragma unroll
  for (int i = 0; i < 8; ++i) a[i] = row[16 * half + i];
#pragma unroll
  for (int i = 0; i < 8; ++i) a[8 + i] = row[16 * half + 8 + i];
  return a;
}

static __device__ inline v16bf frag_A_f32(const float* t, int ld) {
  int lane = threadIdx.x & 31, r = lane & 15, half = lane >> 4;
  const float* row = t + (size_t)r * ld;
  v16bf a;
#pragma unroll
  for (int i = 0; i < 8; ++i) a[i] = f2bf(row[16 * half + i]);
#pragma unroll
  for (int i = 0; i < 8; ++i) a[8 + i] = f2bf(row[16 * half + 8 + i]);
  return a;
}

// A fragment where only rows 0..1 are valid (LSTM batch=2); others zeroed.
static __device__ inline v16bf frag_A2_bf16(const __bf16* h2, int ldh, int koff) {
  int lane = threadIdx.x & 31, r = lane & 15, half = lane >> 4;
  v16bf a;
  if (r < 2) {
    const __bf16* row = h2 + (size_t)r * ldh + koff;
#pragma unroll
    for (int i = 0; i < 8; ++i) a[i] = row[16 * half + i];
#pragma unroll
    for (int i = 0; i < 8; ++i) a[8 + i] = row[16 * half + 8 + i];
  } else {
#pragma unroll
    for (int i = 0; i < 16; ++i) a[i] = f2bf(0.0f);
  }
  return a;
}

static __device__ inline v16bf frag_Brow_bf16(const __bf16* t, int ld) {
  int lane = threadIdx.x & 31, n = lane & 15, half = lane >> 4;
  const __bf16* row = t + (size_t)n * ld + 16 * half;
  v16bf b;
#pragma unroll
  for (int i = 0; i < 16; ++i) b[i] = row[i];
  return b;
}

static __device__ inline v16bf frag_Brow_f32(const float* t, int ld) {
  int lane = threadIdx.x & 31, n = lane & 15, half = lane >> 4;
  const float* row = t + (size_t)n * ld + 16 * half;
  v16bf b;
#pragma unroll
  for (int i = 0; i < 16; ++i) b[i] = f2bf(row[i]);
  return b;
}

// B where source is [k][n] row-major (k varies along ld): used for P@V (V is [key][d])
static __device__ inline v16bf frag_Bk_f32(const float* t, int ld) {
  int lane = threadIdx.x & 31, n = lane & 15, half = lane >> 4;
  v16bf b;
#pragma unroll
  for (int i = 0; i < 16; ++i) b[i] = f2bf(t[(size_t)(16 * half + i) * ld + n]);
  return b;
}

static __device__ inline float redmax16(float v) {
  v = fmaxf(v, __shfl_xor(v, 1, 32));
  v = fmaxf(v, __shfl_xor(v, 2, 32));
  v = fmaxf(v, __shfl_xor(v, 4, 32));
  v = fmaxf(v, __shfl_xor(v, 8, 32));
  return v;
}
static __device__ inline float redsum16(float v) {
  v += __shfl_xor(v, 1, 32);
  v += __shfl_xor(v, 2, 32);
  v += __shfl_xor(v, 4, 32);
  v += __shfl_xor(v, 8, 32);
  return v;
}

// ---------------------------------------------------------------------------
// Tensor Data Mover: async 2D tile load Global -> LDS (CDNA5 ISA ch.8)
// D# group0: count=1 | lds_addr | global_addr[56:0] | type=2
// D# group1: data_size code, tensor dims (large => no OOB clamp), tile dims,
//            48-bit tensor_dim0_stride (row pitch in elements)
// ---------------------------------------------------------------------------
#ifdef HAVE_TDM
static __device__ inline unsigned lds_off(const void* p) {
  return (unsigned)(uintptr_t)p;  // low 32 bits of LDS aperture address
}

static __device__ inline void tdm_load_tile(const void* g, unsigned ldsbyte,
                                            int rows, int cols_elems,
                                            int stride_elems, unsigned dscode) {
  unsigned long long ga = (unsigned long long)(uintptr_t)g;
  v4u g0;
  g0[0] = 1u;                                                  // count=1, user mode
  g0[1] = ldsbyte;                                             // lds_addr (bytes)
  g0[2] = (unsigned)(ga & 0xffffffffu);                        // global_addr[31:0]
  g0[3] = (unsigned)((ga >> 32) & 0x01ffffffu) | 0x80000000u;  // addr[56:32] | type=2
  const unsigned td0 = 1u << 20, td1 = 1u << 20;               // tensor dims (no clamp)
  unsigned long long s0 = (unsigned long long)(unsigned)stride_elems;
  v8i g1;
  g1[0] = (int)(dscode << 16);                                         // data_size
  g1[1] = (int)((td0 & 0xffffu) << 16);                                // tensor_dim0 lo
  g1[2] = (int)(((td0 >> 16) & 0xffffu) | ((td1 & 0xffffu) << 16));    // dim0 hi|dim1 lo
  g1[3] = (int)(((td1 >> 16) & 0xffffu) |
                (((unsigned)cols_elems & 0xffffu) << 16));             // tile_dim0
  g1[4] = (int)((unsigned)rows & 0xffffu);                             // tile_dim1
  g1[5] = (int)(unsigned)(s0 & 0xffffffffu);                           // dim0_stride lo
  g1[6] = (int)(unsigned)((s0 >> 32) & 0xffffu);                       // dim0_stride hi
  g1[7] = 0;
  v4i z4 = {0, 0, 0, 0};
#if defined(__clang_major__) && __clang_major__ >= 23
  v8i z8 = {0, 0, 0, 0, 0, 0, 0, 0};
  __builtin_amdgcn_tensor_load_to_lds(g0, g1, z4, z4, z8, 0);
#else
  __builtin_amdgcn_tensor_load_to_lds(g0, g1, z4, z4, 0);
#endif
}
#endif  // HAVE_TDM

// ---------------------------------------------------------------------------
// bf16-WMMA GEMM: C[M,N](f32) = epilogue(A[M,K](bf16) @ W[N,K](bf16)^T)
// Block tile 64x256, 8 waves (2x4), wave tile 32x64 (2x4 WMMA tiles).
// Double-buffered *bf16* LDS tiles filled by the Tensor Data Mover so the DMA
// for tile k+1 overlaps the WMMAs on tile k; fragments are pure ds_load_b128.
// Requires M%64==0, N%256==0, K%32==0 (true for all call sites here).
// ---------------------------------------------------------------------------
#define GBM 64
#define GBN 256
#define GBK 32
#define GEMM_BUFB ((GBM + GBN) * GBK * 2)      // 20480 bytes per buffer (bf16)
#define GEMM_SMEM (2 * GEMM_BUFB)              // 40960 bytes dynamic LDS

static __device__ inline void gemm_stage(const __bf16* __restrict__ A, int lda, int bm,
                                         const __bf16* __restrict__ W, int ldw, int bn,
                                         int k0, __bf16* sA, __bf16* sB) {
#ifdef HAVE_TDM
  if (threadIdx.x == 0) {
    tdm_load_tile(A + (size_t)bm * lda + k0, lds_off(sA), GBM, GBK, lda, 1u);
    tdm_load_tile(W + (size_t)bn * ldw + k0, lds_off(sB), GBN, GBK, ldw, 1u);
  }
#else
  for (int i = threadIdx.x; i < GBM * GBK; i += 256)
    sA[i] = A[(size_t)(bm + (i >> 5)) * lda + k0 + (i & 31)];
  for (int i = threadIdx.x; i < GBN * GBK; i += 256)
    sB[i] = W[(size_t)(bn + (i >> 5)) * ldw + k0 + (i & 31)];
#endif
}

static __device__ inline void gemm_wait_stage() {
#ifdef HAVE_TDM
  if (threadIdx.x == 0) __builtin_amdgcn_s_wait_tensorcnt(0);
#endif
  __syncthreads();
}

__global__ __launch_bounds__(256) void k_gemm(
    const __bf16* __restrict__ A, int lda,
    const __bf16* __restrict__ W, int ldw,
    float* __restrict__ C, int ldc,
    int M, int N, int K,
    const float* __restrict__ bias,
    const float* __restrict__ rowscale, int rs_stride,
    int accflag) {
  (void)M; (void)N;
  extern __shared__ char gsm[];
  const int tid = threadIdx.x;
  const int wave = tid >> 5;
  const int wm = wave >> 2;  // 0..1
  const int wn = wave & 3;   // 0..3
  const int bm = blockIdx.y * GBM;
  const int bn = blockIdx.x * GBN;

  v8f acc[2][4];
#pragma unroll
  for (int mi = 0; mi < 2; ++mi)
#pragma unroll
    for (int ni = 0; ni < 4; ++ni) acc[mi][ni] = vzero8();

  const int nk = K / GBK;
  gemm_stage(A, lda, bm, W, ldw, bn, 0, (__bf16*)gsm, (__bf16*)(gsm + GBM * GBK * 2));
  int buf = 0;
  for (int ki = 0; ki < nk; ++ki) {
    gemm_wait_stage();  // tile ki resident + visible to all waves
    // LDS pointers via runtime offset arithmetic (pointer arrays constant-fold
    // into addrspacecast static initializers that break ld.lld).
    char* cur = gsm + buf * GEMM_BUFB;
    if (ki + 1 < nk) {  // kick DMA for next tile while we compute
      char* nxt = gsm + (buf ^ 1) * GEMM_BUFB;
      gemm_stage(A, lda, bm, W, ldw, bn, (ki + 1) * GBK,
                 (__bf16*)nxt, (__bf16*)(nxt + GBM * GBK * 2));
    }
    const __bf16* cA = (const __bf16*)cur;
    const __bf16* cB = (const __bf16*)(cur + GBM * GBK * 2);
    v16bf av[2];
#pragma unroll
    for (int mi = 0; mi < 2; ++mi)
      av[mi] = frag_A_bf16(cA + (size_t)(wm * 32 + mi * 16) * GBK, GBK);
#pragma unroll
    for (int ni = 0; ni < 4; ++ni) {
      v16bf bv = frag_Brow_bf16(cB + (size_t)(wn * 64 + ni * 16) * GBK, GBK);
#pragma unroll
      for (int mi = 0; mi < 2; ++mi)
        acc[mi][ni] = WMMA_BF16(av[mi], bv, acc[mi][ni]);
    }
    __syncthreads();
    buf ^= 1;
  }

  const int lane = tid & 31, half = lane >> 4, lc = lane & 15;
#pragma unroll
  for (int mi = 0; mi < 2; ++mi) {
#pragma unroll
    for (int ni = 0; ni < 4; ++ni) {
#pragma unroll
      for (int a = 0; a < 8; ++a) {
        int row = bm + wm * 32 + mi * 16 + a + 8 * half;
        int col = bn + wn * 64 + ni * 16 + lc;
        float v = acc[mi][ni][a];
        if (bias) v += bias[col];
        if (rowscale) v *= rowscale[(size_t)row * rs_stride];
        size_t idx = (size_t)row * ldc + col;
        if (accflag) C[idx] += v; else C[idx] = v;
      }
    }
  }
}

// ---------------------------------------------------------------------------
// fp32 -> bf16 bulk convert (8 elems/thread, b128 in / b128 out)
// ---------------------------------------------------------------------------
__global__ __launch_bounds__(256) void k_f32tobf16(const float* __restrict__ s,
                                                   __bf16* __restrict__ d,
                                                   long long n) {
  long long i = ((long long)blockIdx.x * 256 + threadIdx.x) * 8;
  if (i + 8 > n) return;
  float4 f0 = *(const float4*)(s + i);
  float4 f1 = *(const float4*)(s + i + 4);
  union { __bf16 h[8]; uint4 u; } cv;
  cv.h[0] = f2bf(f0.x); cv.h[1] = f2bf(f0.y); cv.h[2] = f2bf(f0.z); cv.h[3] = f2bf(f0.w);
  cv.h[4] = f2bf(f1.x); cv.h[5] = f2bf(f1.y); cv.h[6] = f2bf(f1.z); cv.h[7] = f2bf(f1.w);
  *(uint4*)(d + i) = cv.u;
}

// ---------------------------------------------------------------------------
// RMSNorm: one block per row; emits bf16 (GEMM operand) + optional fp32 copy
// ---------------------------------------------------------------------------
__global__ __launch_bounds__(256) void k_rmsnorm(const float* __restrict__ x,
                                                 const float* __restrict__ w,
                                                 __bf16* __restrict__ obf,
                                                 float* __restrict__ of32, int Hn) {
  __shared__ float red[256];
  const int row = blockIdx.x;
  const float* xr = x + (size_t)row * Hn;
  float s = 0.0f;
  for (int i = threadIdx.x; i < Hn; i += 256) { float v = xr[i]; s += v * v; }
  red[threadIdx.x] = s;
  __syncthreads();
  for (int st = 128; st > 0; st >>= 1) {
    if (threadIdx.x < st) red[threadIdx.x] += red[threadIdx.x + st];
    __syncthreads();
  }
  const float inv = rsqrtf(red[0] / (float)Hn + 1e-6f);
  for (int i = threadIdx.x; i < Hn; i += 256) {
    float v = xr[i] * inv * w[i];
    obf[(size_t)row * Hn + i] = f2bf(v);
    if (of32) of32[(size_t)row * Hn + i] = v;
  }
}

// ---------------------------------------------------------------------------
// Persistent cooperative LSTM. 32 blocks x 256 thr. Block b owns 32 cells.
// Whh slice (4 gates x 32 cells = 128 rows x 1024 K) cached as bf16 in 256KB
// LDS for the whole sequence. One grid barrier per timestep. hs emitted bf16.
// ---------------------------------------------------------------------------
static __device__ inline void gbar(unsigned* c, unsigned n) {
  __syncthreads();
  __threadfence();
  if (threadIdx.x == 0) {
    atomicAdd(c, 1u);
    while (atomicAdd(c, 0u) < n) __builtin_amdgcn_s_sleep(1);
  }
  __threadfence();
  __syncthreads();
}

#define LSTM_NBLK 32
#define LSTM_SMEM (262144 + 4096 + 1024 + 256)

__global__ __launch_bounds__(256) void k_lstm(
    const float* __restrict__ Xp,   // [2*1024, 4096]
    const float* __restrict__ whh,  // [4096, 1024]
    __bf16* __restrict__ hs,        // [2*1024, 1024] out (bf16, feeds QKV GEMMs)
    float* __restrict__ hbuf,       // [2][2*1024] global h double-buffer (pre-zeroed)
    unsigned* __restrict__ cnt) {   // [1024] barrier counters (pre-zeroed)
  extern __shared__ char smem[];
  __bf16* swhh = (__bf16*)smem;                         // 128*1024 bf16 = 256KB
  __bf16* sh   = (__bf16*)(smem + 262144);              // 2*1024 bf16
  float*  sg   = (float*)(smem + 262144 + 4096);        // 2*128 f32
  float*  sc   = (float*)(smem + 262144 + 4096 + 1024); // 2*32 f32

  const int tid = threadIdx.x;
  const int wave = tid >> 5;
  const int lane = tid & 31, half = lane >> 4, lc = lane & 15;
  const int cellbase = blockIdx.x * 32;

  // cache this block's Whh slice as bf16 (vectorized: float4 -> 4x bf16)
  for (int idx = tid; idx < 32768; idx += 256) {
    int lc2 = idx >> 8;             // local column (gate*32 + cell)
    int kq = (idx & 255) * 4;       // k quad
    int grow = ((lc2 >> 5) << 10) + cellbase + (lc2 & 31);
    float4 f = *(const float4*)(whh + (size_t)grow * 1024 + kq);
    union { __bf16 h[4]; uint2 u; } cv;
    cv.h[0] = f2bf(f.x); cv.h[1] = f2bf(f.y); cv.h[2] = f2bf(f.z); cv.h[3] = f2bf(f.w);
    ((uint2*)swhh)[idx] = cv.u;
  }
  if (tid < 64) sc[tid] = 0.0f;
  __syncthreads();

  for (int t = 0; t < 1024; ++t) {
    // stage h_prev as bf16: 8 elems/thread, one b128 LDS store
    {
      const float4* src = (const float4*)(hbuf + (size_t)(t & 1) * 2048);
      float4 f0 = src[tid * 2];
      float4 f1 = src[tid * 2 + 1];
      union { __bf16 h[8]; uint4 u; } cv;
      cv.h[0] = f2bf(f0.x); cv.h[1] = f2bf(f0.y); cv.h[2] = f2bf(f0.z); cv.h[3] = f2bf(f0.w);
      cv.h[4] = f2bf(f1.x); cv.h[5] = f2bf(f1.y); cv.h[6] = f2bf(f1.z); cv.h[7] = f2bf(f1.w);
      ((uint4*)sh)[tid] = cv.u;
    }
    __syncthreads();

    // g_tile = h_prev @ Whh_slice^T   (one 16x16 WMMA tile per wave, rows 0-1 valid)
    v8f acc = vzero8();
    const __bf16* wbase = swhh + (size_t)(wave * 16) * 1024;
#pragma unroll 4
    for (int kk = 0; kk < 32; ++kk) {
      v16bf a = frag_A2_bf16(sh, 1024, kk * 32);
      v16bf b = frag_Brow_bf16(wbase + kk * 32, 1024);
      acc = WMMA_BF16(a, b, acc);
    }
    if (half == 0) {                  // rows 0 (acc[0]) and 1 (acc[1]) live in lanes 0-15
      sg[0 * 128 + wave * 16 + lc] = acc[0];
      sg[1 * 128 + wave * 16 + lc] = acc[1];
    }
    __syncthreads();

    // gate nonlinearity + state update for our 32 cells, both batches
    if (tid < 64) {
      int b2 = tid >> 5, ci = tid & 31;
      size_t xrow = ((size_t)b2 * 1024 + t) * 4096;
      int cell = cellbase + ci;
      float gi = sg[b2 * 128 + 0  + ci] + Xp[xrow + 0    + cell];
      float gf = sg[b2 * 128 + 32 + ci] + Xp[xrow + 1024 + cell];
      float gg = sg[b2 * 128 + 64 + ci] + Xp[xrow + 2048 + cell];
      float go = sg[b2 * 128 + 96 + ci] + Xp[xrow + 3072 + cell];
      float cn = sigmoidf_(gf) * sc[tid] + sigmoidf_(gi) * tanhf(gg);
      float hn = sigmoidf_(go) * tanhf(cn);
      sc[tid] = cn;
      hbuf[(size_t)((t + 1) & 1) * 2048 + (size_t)b2 * 1024 + cell] = hn;
      hs[((size_t)b2 * 1024 + t) * 1024 + cell] = f2bf(hn);
    }
    gbar(cnt + t, LSTM_NBLK);
  }
}

// ---------------------------------------------------------------------------
// RoPE on q and k in place. One thread per (token, head, pair d<32).
// ---------------------------------------------------------------------------
__global__ __launch_bounds__(256) void k_rope(float* __restrict__ q, float* __restrict__ k,
                                              const float* __restrict__ cosb,
                                              const float* __restrict__ sinb) {
  size_t i = (size_t)blockIdx.x * blockDim.x + threadIdx.x;
  if (i >= (size_t)2048 * 16 * 32) return;
  int d = (int)(i & 31);
  size_t r = i >> 5;
  int hh = (int)(r & 15);
  size_t tok = r >> 4;  // b*S+s
  size_t base = tok * 1024 + (size_t)hh * 64;
  float c1 = cosb[tok * 64 + d],      s1 = sinb[tok * 64 + d];
  float c2 = cosb[tok * 64 + 32 + d], s2 = sinb[tok * 64 + 32 + d];
  float q1 = q[base + d], q2 = q[base + 32 + d];
  q[base + d]      = q1 * c1 - q2 * s1;
  q[base + 32 + d] = q2 * c2 + q1 * s2;
  float k1 = k[base + d], k2 = k[base + 32 + d];
  k[base + d]      = k1 * c1 - k2 * s1;
  k[base + 32 + d] = k2 * c2 + k1 * s2;
}

// ---------------------------------------------------------------------------
// Causal flash attention. One wave per (b, head, 16-query tile).
// Scores and P@V via bf16 WMMA; online softmax in fp32. Output bf16.
// ---------------------------------------------------------------------------
__global__ __launch_bounds__(256) void k_attn(const float* __restrict__ Q,
                                              const float* __restrict__ Kt,
                                              const float* __restrict__ V,
                                              __bf16* __restrict__ O) {
  __shared__ __bf16 sP[8][16 * 32];
  const int wave = threadIdx.x >> 5;
  const int lane = threadIdx.x & 31, half = lane >> 4, lc = lane & 15;
  const int wid = blockIdx.x * 8 + wave;
  const int qt = wid & 63;
  const int bh = wid >> 6;
  const int h = bh & 15, b = bh >> 4;
  const int q0 = qt * 16;
  const int ld = 1024;
  const float SCALING = 0.125f;  // 64^-0.5
  const float* Qp = Q + (size_t)b * 1024 * ld + h * 64;
  const float* Kp = Kt + (size_t)b * 1024 * ld + h * 64;
  const float* Vp = V + (size_t)b * 1024 * ld + h * 64;

  v16bf qa[2];
  qa[0] = frag_A_f32(Qp + (size_t)q0 * ld, ld);
  qa[1] = frag_A_f32(Qp + (size_t)q0 * ld + 32, ld);

  float mrun[8], lrun[8];
  v8f o[4];
#pragma unroll
  for (int a = 0; a < 8; ++a) { mrun[a] = -1e30f; lrun[a] = 0.0f; }
#pragma unroll
  for (int ni = 0; ni < 4; ++ni) o[ni] = vzero8();

  for (int kc = 0; kc < q0 + 16; kc += 32) {
    v8f s0 = vzero8(), s1 = vzero8();
#pragma unroll
    for (int c = 0; c < 2; ++c) {
      v16bf kb0 = frag_Brow_f32(Kp + (size_t)(kc)      * ld + c * 32, ld);
      v16bf kb1 = frag_Brow_f32(Kp + (size_t)(kc + 16) * ld + c * 32, ld);
      s0 = WMMA_BF16(qa[c], kb0, s0);
      s1 = WMMA_BF16(qa[c], kb1, s1);
    }
#pragma unroll
    for (int a = 0; a < 8; ++a) {
      int row = q0 + a + 8 * half;
      float v0 = s0[a] * SCALING; if (kc + lc      > row) v0 = -1e30f;
      float v1 = s1[a] * SCALING; if (kc + 16 + lc > row) v1 = -1e30f;
      float mx = redmax16(fmaxf(v0, v1));
      float mnew = fmaxf(mrun[a], mx);
      float alpha = __expf(mrun[a] - mnew);
      float p0 = __expf(v0 - mnew);
      float p1 = __expf(v1 - mnew);
      lrun[a] = lrun[a] * alpha + redsum16(p0 + p1);
      mrun[a] = mnew;
#pragma unroll
      for (int ni = 0; ni < 4; ++ni) o[ni][a] *= alpha;
      sP[wave][(a + 8 * half) * 32 + lc]      = f2bf(p0);
      sP[wave][(a + 8 * half) * 32 + 16 + lc] = f2bf(p1);
    }
    // wave-private LDS round-trip to convert C-layout P into A-layout fragment
    v16bf pa = frag_A_bf16(&sP[wave][0], 32);
#pragma unroll
    for (int ni = 0; ni < 4; ++ni) {
      v16bf vb = frag_Bk_f32(Vp + (size_t)kc * ld + ni * 16, ld);
      o[ni] = WMMA_BF16(pa, vb, o[ni]);
    }
  }
#pragma unroll
  for (int ni = 0; ni < 4; ++ni)
#pragma unroll
    for (int a = 0; a < 8; ++a) {
      int row = q0 + a + 8 * half;
      int d = ni * 16 + lc;
      O[((size_t)b * 1024 + row) * ld + h * 64 + d] = f2bf(o[ni][a] / lrun[a]);
    }
}

// ---------------------------------------------------------------------------
// Small elementwise kernels
// ---------------------------------------------------------------------------
__global__ void k_vecadd(const float* a, const float* b, float* o, int n) {
  int i = blockIdx.x * blockDim.x + threadIdx.x;
  if (i < n) o[i] = a[i] + b[i];
}

// o = bf16(silu(t1) * t3)
__global__ void k_silu_mul(const float* __restrict__ t1, const float* __restrict__ t3,
                           __bf16* __restrict__ o, int n) {
  int i = blockIdx.x * blockDim.x + threadIdx.x;
  if (i < n) { float a = t1[i]; o[i] = f2bf(a / (1.0f + __expf(-a)) * t3[i]); }
}

// Top-1 gate: softmax over 3 logits; select argmax(score+bias); weight=score (no bias)
__global__ void k_gate(const float* __restrict__ xn, const float* __restrict__ gw,
                       const float* __restrict__ gb, float* __restrict__ wtok) {
  int t = blockIdx.x * blockDim.x + threadIdx.x;
  if (t >= 2048) return;
  const float* xr = xn + (size_t)t * 1024;
  float lg[3];
#pragma unroll
  for (int e = 0; e < 3; ++e) {
    float s = 0.0f;
    for (int k = 0; k < 1024; ++k) s += xr[k] * gw[e * 1024 + k];
    lg[e] = s;
  }
  float mx = fmaxf(lg[0], fmaxf(lg[1], lg[2]));
  float ex[3], den = 0.0f;
#pragma unroll
  for (int e = 0; e < 3; ++e) { ex[e] = __expf(lg[e] - mx); den += ex[e]; }
  float gs[3];
#pragma unroll
  for (int e = 0; e < 3; ++e) gs[e] = ex[e] / den;
  int arg = 0; float best = gs[0] + gb[0];
#pragma unroll
  for (int e = 1; e < 3; ++e) { float v = gs[e] + gb[e]; if (v > best) { best = v; arg = e; } }
#pragma unroll
  for (int e = 0; e < 3; ++e) wtok[(size_t)t * 3 + e] = (e == arg) ? gs[e] : 0.0f;
}

// ---------------------------------------------------------------------------
// Orchestration
// ---------------------------------------------------------------------------
extern "C" void kernel_launch(void* const* d_in, const int* in_sizes, int n_in,
                              void* d_out, int out_size, void* d_ws, size_t ws_size,
                              hipStream_t stream) {
  (void)in_sizes; (void)n_in; (void)out_size; (void)ws_size;
  const float* hidden = (const float*)d_in[0];
  const float* cosb   = (const float*)d_in[1];
  const float* sinb   = (const float*)d_in[2];
  /* d_in[3] attention_mask: causal, handled analytically */
  const float* n1w = (const float*)d_in[4];
  const float* n2w = (const float*)d_in[5];
  const float* wih = (const float*)d_in[6];
  const float* whh = (const float*)d_in[7];
  const float* bih = (const float*)d_in[8];
  const float* bhh = (const float*)d_in[9];
  const float* wq  = (const float*)d_in[10];
  const float* bq  = (const float*)d_in[11];
  const float* wk  = (const float*)d_in[12];
  const float* bk  = (const float*)d_in[13];
  const float* wv  = (const float*)d_in[14];
  const float* bv  = (const float*)d_in[15];
  const float* wo  = (const float*)d_in[16];
  const float* gw  = (const float*)d_in[17];
  const float* gb  = (const float*)d_in[18];
  const float* ew1 = (const float*)d_in[19];
  const float* ew2 = (const float*)d_in[20];
  const float* ew3 = (const float*)d_in[21];
  const float* sw1 = (const float*)d_in[22];
  const float* sw2 = (const float*)d_in[23];
  const float* sw3 = (const float*)d_in[24];
  float* out = (float*)d_out;

  const size_t T = 2048;               // B*S tokens
  const size_t TH = T * 1024 * 4;      // 8MB (fp32 token matrix)
  char* p = (char*)d_ws;
  size_t off = 0;
  auto bump = [&](size_t bytes) -> char* {
    char* r = p + off;
    off += (bytes + 255) & ~(size_t)255;
    return r;
  };
  unsigned* cnt = (unsigned*)bump(4096);           // 1024 barrier counters
  float* hbuf   = (float*)bump(4 * 4096);          // h double buffer [2][2048]
  float* bsum   = (float*)bump(4096 * 4);
  float* wtok   = (float*)bump(T * 3 * 4);
  float* Xp     = (float*)bump(T * 4096 * 4);      // 32MB; reused as t1|t3 for MoE
  float* qb     = (float*)bump(TH);                // q; reused as fp32 xn2 later
  float* kb     = (float*)bump(TH);
  float* vb     = (float*)bump(TH);
  float* x2     = (float*)bump(TH);
  // bf16 operand pool
  __bf16* wihb = (__bf16*)bump((size_t)4096 * 1024 * 2);
  __bf16* wqb  = (__bf16*)bump((size_t)1024 * 1024 * 2);
  __bf16* wkb  = (__bf16*)bump((size_t)1024 * 1024 * 2);
  __bf16* wvb  = (__bf16*)bump((size_t)1024 * 1024 * 2);
  __bf16* wob  = (__bf16*)bump((size_t)1024 * 1024 * 2);
  __bf16* ew1b = (__bf16*)bump((size_t)3 * 2048 * 1024 * 2);
  __bf16* ew2b = (__bf16*)bump((size_t)3 * 1024 * 2048 * 2);
  __bf16* ew3b = (__bf16*)bump((size_t)3 * 2048 * 1024 * 2);
  __bf16* sw1b = (__bf16*)bump((size_t)2048 * 1024 * 2);
  __bf16* sw2b = (__bf16*)bump((size_t)1024 * 2048 * 2);
  __bf16* sw3b = (__bf16*)bump((size_t)2048 * 1024 * 2);
  __bf16* xnb  = (__bf16*)bump(T * 1024 * 2);      // xn1 then xn2 (bf16)
  __bf16* hsb  = (__bf16*)bump(T * 1024 * 2);      // LSTM h sequence (bf16)
  __bf16* atb  = (__bf16*)bump(T * 1024 * 2);      // attention output (bf16)
  __bf16* t1b  = (__bf16*)bump(T * 2048 * 2);      // silu(t1)*t3 (bf16)
  float* t1 = Xp;
  float* t3 = Xp + T * 2048;
  float* xn2f = qb;  // fp32 xn2 for the router (q dead after attention)

  // zero barrier counters + h state (cnt and hbuf are contiguous: 4096+16384)
  (void)hipMemsetAsync(cnt, 0, 4096 + 16384, stream);

  // one-time weight conversions fp32 -> bf16 (halves GEMM operand traffic)
  auto cvt = [&](const float* s, __bf16* d, long long n) {
    k_f32tobf16<<<(unsigned)(n / 2048), 256, 0, stream>>>(s, d, n);
  };
  cvt(wih, wihb, (long long)4096 * 1024);
  cvt(wq, wqb, (long long)1024 * 1024);
  cvt(wk, wkb, (long long)1024 * 1024);
  cvt(wv, wvb, (long long)1024 * 1024);
  cvt(wo, wob, (long long)1024 * 1024);
  cvt(ew1, ew1b, (long long)3 * 2048 * 1024);
  cvt(ew2, ew2b, (long long)3 * 1024 * 2048);
  cvt(ew3, ew3b, (long long)3 * 2048 * 1024);
  cvt(sw1, sw1b, (long long)2048 * 1024);
  cvt(sw2, sw2b, (long long)1024 * 2048);
  cvt(sw3, sw3b, (long long)2048 * 1024);

  // bsum = bih + bhh
  k_vecadd<<<16, 256, 0, stream>>>(bih, bhh, bsum, 4096);

  // xn1 = rmsnorm(hidden) -> bf16
  k_rmsnorm<<<(int)T, 256, 0, stream>>>(hidden, n1w, xnb, nullptr, 1024);

  // Xp = xn1 @ Wih^T + bsum          [2048 x 4096]
  k_gemm<<<dim3(16, 32), 256, GEMM_SMEM, stream>>>(xnb, 1024, wihb, 1024, Xp, 4096,
                                                   2048, 4096, 1024, bsum, nullptr, 0, 0);

  // LSTM recurrence (persistent cooperative, Whh in LDS)
  k_lstm<<<LSTM_NBLK, 256, LSTM_SMEM, stream>>>(Xp, whh, hsb, hbuf, cnt);

  // QKV projections
  k_gemm<<<dim3(4, 32), 256, GEMM_SMEM, stream>>>(hsb, 1024, wqb, 1024, qb, 1024,
                                                  2048, 1024, 1024, bq, nullptr, 0, 0);
  k_gemm<<<dim3(4, 32), 256, GEMM_SMEM, stream>>>(hsb, 1024, wkb, 1024, kb, 1024,
                                                  2048, 1024, 1024, bk, nullptr, 0, 0);
  k_gemm<<<dim3(4, 32), 256, GEMM_SMEM, stream>>>(hsb, 1024, wvb, 1024, vb, 1024,
                                                  2048, 1024, 1024, bv, nullptr, 0, 0);

  // RoPE in place on q,k
  k_rope<<<4096, 256, 0, stream>>>(qb, kb, cosb, sinb);

  // causal flash attention -> atb (bf16, [token, h*64+d])
  k_attn<<<256, 256, 0, stream>>>(qb, kb, vb, atb);

  // x2 = hidden + atb @ wo^T
  (void)hipMemcpyAsync(x2, hidden, TH, hipMemcpyDeviceToDevice, stream);
  k_gemm<<<dim3(4, 32), 256, GEMM_SMEM, stream>>>(atb, 1024, wob, 1024, x2, 1024,
                                                  2048, 1024, 1024, nullptr, nullptr, 0, 1);

  // xn2 = rmsnorm(x2) -> bf16 (GEMMs) + fp32 (router); gating
  k_rmsnorm<<<(int)T, 256, 0, stream>>>(x2, n2w, xnb, xn2f, 1024);
  k_gate<<<8, 256, 0, stream>>>(xn2f, gw, gb, wtok);

  // out = x2 (residual) + sum_e we*expert_e(xn2) + shared(xn2)
  (void)hipMemcpyAsync(out, x2, TH, hipMemcpyDeviceToDevice, stream);

  for (int e = 0; e < 3; ++e) {
    const __bf16* w1 = ew1b + (size_t)e * 2048 * 1024;
    const __bf16* w3 = ew3b + (size_t)e * 2048 * 1024;
    const __bf16* w2 = ew2b + (size_t)e * 1024 * 2048;
    k_gemm<<<dim3(8, 32), 256, GEMM_SMEM, stream>>>(xnb, 1024, w1, 1024, t1, 2048,
                                                    2048, 2048, 1024, nullptr, nullptr, 0, 0);
    k_gemm<<<dim3(8, 32), 256, GEMM_SMEM, stream>>>(xnb, 1024, w3, 1024, t3, 2048,
                                                    2048, 2048, 1024, nullptr, nullptr, 0, 0);
    k_silu_mul<<<16384, 256, 0, stream>>>(t1, t3, t1b, 2048 * 2048);
    k_gemm<<<dim3(4, 32), 256, GEMM_SMEM, stream>>>(t1b, 2048, w2, 2048, out, 1024,
                                                    2048, 1024, 2048, nullptr, wtok + e, 3, 1);
  }
  // shared expert (weight 1.0)
  k_gemm<<<dim3(8, 32), 256, GEMM_SMEM, stream>>>(xnb, 1024, sw1b, 1024, t1, 2048,
                                                  2048, 2048, 1024, nullptr, nullptr, 0, 0);
  k_gemm<<<dim3(8, 32), 256, GEMM_SMEM, stream>>>(xnb, 1024, sw3b, 1024, t3, 2048,
                                                  2048, 2048, 1024, nullptr, nullptr, 0, 0);
  k_silu_mul<<<16384, 256, 0, stream>>>(t1, t3, t1b, 2048 * 2048);
  k_gemm<<<dim3(4, 32), 256, GEMM_SMEM, stream>>>(t1b, 2048, sw2b, 2048, out, 1024,
                                                  2048, 1024, 2048, nullptr, nullptr, 0, 1);
}
